// OldODEFunc_19009525252585
// MI455X (gfx1250) — matrix operand order repaired and checked by reading, not compiled, not run on hardware
//
#include <hip/hip_runtime.h>

// ---------------- problem constants ----------------
#define B_DIM   8192          // batch
#define S_DIM   268           // state width
#define H_DIM   1024          // hidden
#define A_DIM   256           // h_part width
#define Z_DIM   8
#define E_DIM   4
#define K0_PAD  288           // 270 padded to multiple of 32
#define NF_PAD  384           // 268 padded to multiple of 128

// ---------------- vector types ----------------
typedef __attribute__((ext_vector_type(16))) __bf16         v16bf;
typedef __attribute__((ext_vector_type(8)))  float          v8f;
typedef __attribute__((ext_vector_type(8)))  unsigned short us8;
typedef __attribute__((ext_vector_type(16))) unsigned short us16;

// ---------------- bf16 helpers ----------------
__device__ __forceinline__ unsigned short f2bf(float f) {
  unsigned int x = __builtin_bit_cast(unsigned int, f);
  x = (x + 0x7FFFu + ((x >> 16) & 1u)) >> 16;   // round-to-nearest-even
  return (unsigned short)x;
}
__device__ __forceinline__ float bf2f(unsigned short u) {
  unsigned int x = ((unsigned int)u) << 16;
  return __builtin_bit_cast(float, x);
}

// ---------------- WMMA GEMM ----------------
// C[M x N] = act( A[M x K](bf16,row-major) * W[N x K](bf16,row-major)^T + bias )
// workgroup tile 256x128, 8 waves (wave32) as 4(M) x 2(N); wave tile 64x64
// = 4x4 WMMA 16x16x32 accumulators -> 16 WMMAs per K-slice per wave.
// K staged through LDS in 32-deep slices; next slice software-pipelined into
// registers while the WMMA chain runs.
#define LDSS 48   // LDS row stride in bf16 elems (96B: b128-aligned, staggers banks)

__device__ __forceinline__ v8f wmma_bf16(v16bf a, v16bf b, v8f c) {
  return __builtin_amdgcn_wmma_f32_16x16x32_bf16(false, a, false, b, (short)0, c, false, false);
}

// A-matrix fragment (16x32 bf16): lane half h, row = lane&15.
// element j<8  -> K = h*8 + j        (VGPR0-3)
// element j>=8 -> K = 16 + h*8 + j-8 (VGPR4-7)
__device__ __forceinline__ v16bf frag_ld_a(const unsigned short* s, int rowBase, int lane) {
  const int h = lane >> 4;
  const unsigned short* p = s + (rowBase + (lane & 15)) * LDSS + h * 8;
  us8 lo = *(const us8*)p;
  us8 hi = *(const us8*)(p + 16);
  us16 u = __builtin_shufflevector(lo, hi, 0,1,2,3,4,5,6,7,8,9,10,11,12,13,14,15);
  return __builtin_bit_cast(v16bf, u);
}
// B-matrix fragment (32x16 bf16) from W rows (N x K): lane half h,
// col n = lane&15, element j<8 -> K = h*8 + j, j>=8 -> K = 16 + h*8 + j-8
__device__ __forceinline__ v16bf frag_ld_b(const unsigned short* s, int rowBase, int lane) {
  const int h = lane >> 4;
  const unsigned short* p = s + (rowBase + (lane & 15)) * LDSS + h * 8;
  us8 lo = *(const us8*)p;
  us8 hi = *(const us8*)(p + 16);
  us16 u = __builtin_shufflevector(lo, hi, 0,1,2,3,4,5,6,7,8,9,10,11,12,13,14,15);
  return __builtin_bit_cast(v16bf, u);
}

// MODE: 0=relu->bf16  1=tanh->bf16  2=tanh(acc+bias+res)->bf16  3=plain->f32  4=plain->bf16
template<int MODE>
__global__ __launch_bounds__(256, 2)
void gemm_wmma_bf16(const unsigned short* __restrict__ A, int lda,
                    const unsigned short* __restrict__ W, int ldw,
                    const float* __restrict__ bias,
                    const unsigned short* __restrict__ Res, int ldres,
                    unsigned short* __restrict__ outB,
                    float* __restrict__ outF, int ldo,
                    int K) {
  __shared__ unsigned short As[256 * LDSS];   // 24 KB
  __shared__ unsigned short Bs[128 * LDSS];   // 12 KB

  const int bN   = blockIdx.x * 128;
  const int bM   = blockIdx.y * 256;
  const int t    = threadIdx.x;
  const int lane = t & 31;
  const int wave = t >> 5;
  const int wm   = (wave >> 1) * 64;   // wave M offset in tile (0,64,128,192)
  const int wn   = (wave & 1) * 64;    // wave N offset in tile (0,64)
  const int cr   = t >> 2;             // 0..63 staging row base
  const int cc   = (t & 3) * 8;        // 0,8,16,24 staging col (bf16 elems)

  const v8f vzero = {0.f,0.f,0.f,0.f,0.f,0.f,0.f,0.f};
  v8f acc[4][4];
#pragma unroll
  for (int i = 0; i < 4; ++i)
#pragma unroll
    for (int j = 0; j < 4; ++j) acc[i][j] = vzero;

  const unsigned short* Ap = A + (size_t)bM * lda + cc;
  const unsigned short* Wp = W + (size_t)bN * ldw + cc;

  us8 ar[4], wr[2];
  // prefetch K-slice 0 into registers
#pragma unroll
  for (int p = 0; p < 4; ++p) ar[p] = *(const us8*)(Ap + (size_t)(cr + p * 64) * lda);
#pragma unroll
  for (int p = 0; p < 2; ++p) wr[p] = *(const us8*)(Wp + (size_t)(cr + p * 64) * ldw);

  for (int k0 = 0; k0 < K; k0 += 32) {
    __syncthreads();                          // prior slice's frag reads done
#pragma unroll
    for (int p = 0; p < 4; ++p) *(us8*)(As + (cr + p * 64) * LDSS + cc) = ar[p];
#pragma unroll
    for (int p = 0; p < 2; ++p) *(us8*)(Bs + (cr + p * 64) * LDSS + cc) = wr[p];
    __syncthreads();                          // slice visible

    // software pipeline: issue next slice's global loads before the WMMA chain
    const int kn = (k0 + 32 < K) ? (k0 + 32) : k0;   // last iter: harmless refetch
#pragma unroll
    for (int p = 0; p < 4; ++p) ar[p] = *(const us8*)(Ap + (size_t)(cr + p * 64) * lda + kn);
#pragma unroll
    for (int p = 0; p < 2; ++p) wr[p] = *(const us8*)(Wp + (size_t)(cr + p * 64) * ldw + kn);

    v16bf af[4], bfr[4];
#pragma unroll
    for (int i = 0; i < 4; ++i) af[i]  = frag_ld_a(As, wm + i * 16, lane);
#pragma unroll
    for (int j = 0; j < 4; ++j) bfr[j] = frag_ld_b(Bs, wn + j * 16, lane);
#pragma unroll
    for (int i = 0; i < 4; ++i)
#pragma unroll
      for (int j = 0; j < 4; ++j)
        acc[i][j] = wmma_bf16(af[i], bfr[j], acc[i][j]);
  }

  // epilogue: C/D layout -> element v of v8f: row = v + (lane>=16?8:0), col = lane&15
  const int h  = lane >> 4;
  const int cl = lane & 15;
#pragma unroll
  for (int i = 0; i < 4; ++i) {
#pragma unroll
    for (int j = 0; j < 4; ++j) {
      const int col = bN + wn + j * 16 + cl;
      const float bv = bias[col];
#pragma unroll
      for (int v = 0; v < 8; ++v) {
        const int row = bM + wm + i * 16 + h * 8 + v;
        float x = acc[i][j][v] + bv;
        if (MODE == 0)      x = fmaxf(x, 0.0f);
        else if (MODE == 1) x = tanhf(x);
        else if (MODE == 2) x = tanhf(x + bf2f(Res[(size_t)row * ldres + col]));
        if (MODE == 3) outF[(size_t)row * ldo + col] = x;
        else           outB[(size_t)row * ldo + col] = f2bf(x);
      }
    }
  }
}

// ---------------- prep / epilogue kernels ----------------
__global__ void k_cast(const float* __restrict__ s, unsigned short* __restrict__ d, long long n) {
  long long i = (long long)blockIdx.x * blockDim.x + threadIdx.x;
  if (i < n) d[i] = f2bf(s[i]);
}
__global__ void k_pad2d(const float* __restrict__ s, unsigned short* __restrict__ d,
                        int srcR, int srcC, int dstR, int dstC) {
  long long i = (long long)blockIdx.x * blockDim.x + threadIdx.x;
  long long n = (long long)dstR * dstC;
  if (i >= n) return;
  int r = (int)(i / dstC), c = (int)(i % dstC);
  float v = (r < srcR && c < srcC) ? s[(long long)r * srcC + c] : 0.0f;
  d[i] = f2bf(v);
}
__global__ void k_slice2d(const float* __restrict__ s, unsigned short* __restrict__ d,
                          int rows, int srcC, int colOff, int dstC) {
  long long i = (long long)blockIdx.x * blockDim.x + threadIdx.x;
  long long n = (long long)rows * dstC;
  if (i >= n) return;
  int r = (int)(i / dstC), c = (int)(i % dstC);
  d[i] = f2bf(s[(long long)r * srcC + colOff + c]);
}
__global__ void k_padbias(const float* __restrict__ s, float* __restrict__ d, int n, int nt) {
  int i = blockIdx.x * blockDim.x + threadIdx.x;
  if (i < nt) d[i] = (i < n) ? s[i] : 0.0f;
}
__global__ void k_buildx(const float* __restrict__ state, const float* __restrict__ tptr,
                         unsigned short* __restrict__ X) {
  long long i = (long long)blockIdx.x * blockDim.x + threadIdx.x;
  long long n = (long long)B_DIM * K0_PAD;
  if (i >= n) return;
  int m = (int)(i / K0_PAD), k = (int)(i % K0_PAD);
  float v;
  const float PI_OVER_12 = 0.26179938779914943654f;   // 2*pi/24
  if (k < S_DIM)           v = state[(long long)m * S_DIM + k];
  else if (k == S_DIM)     v = __sinf(tptr[0] * PI_OVER_12);
  else if (k == S_DIM + 1) v = __cosf(tptr[0] * PI_OVER_12);
  else                     v = 0.0f;
  X[i] = f2bf(v);
}
// tiny loc path: per batch row, 8->4->4->4->8 dense chain in f32 VALU
__global__ void k_loc(const float* __restrict__ state,
                      const float* __restrict__ lp_in_w,  const float* __restrict__ lp_in_b,
                      const float* __restrict__ lp_out_w, const float* __restrict__ lp_out_b,
                      const float* __restrict__ loc_proj_w, const float* __restrict__ loc_proj_b,
                      const float* __restrict__ loc_back_w, const float* __restrict__ loc_back_b,
                      float* __restrict__ DL) {
  int m = blockIdx.x * blockDim.x + threadIdx.x;
  if (m >= B_DIM) return;
  const float* loc = state + (size_t)m * S_DIM + A_DIM;
  float lp[E_DIM], vv[E_DIM], enh[E_DIM], dd[E_DIM];
#pragma unroll
  for (int e = 0; e < E_DIM; ++e) {
    float a = loc_proj_b[e];
#pragma unroll
    for (int z = 0; z < Z_DIM; ++z) a += loc_proj_w[e * Z_DIM + z] * loc[z];
    lp[e] = a;
  }
#pragma unroll
  for (int e = 0; e < E_DIM; ++e) {
    float a = lp_in_b[2 * E_DIM + e];
#pragma unroll
    for (int j = 0; j < E_DIM; ++j) a += lp_in_w[(2 * E_DIM + e) * E_DIM + j] * lp[j];
    vv[e] = a;
  }
#pragma unroll
  for (int e = 0; e < E_DIM; ++e) {
    float a = lp_out_b[e];
#pragma unroll
    for (int j = 0; j < E_DIM; ++j) a += lp_out_w[e * E_DIM + j] * vv[j];
    enh[e] = a;
  }
#pragma unroll
  for (int e = 0; e < E_DIM; ++e) dd[e] = enh[e] - lp[e];
#pragma unroll
  for (int z = 0; z < Z_DIM; ++z) {
    float a = loc_back_b[z];
#pragma unroll
    for (int e = 0; e < E_DIM; ++e) a += loc_back_w[z * E_DIM + e] * dd[e];
    DL[(size_t)m * Z_DIM + z] = a;
  }
}
__global__ void k_combine(const float* __restrict__ coreF, const float* __restrict__ EH,
                          const float* __restrict__ state, const float* __restrict__ DL,
                          float* __restrict__ out) {
  long long i = (long long)blockIdx.x * blockDim.x + threadIdx.x;
  long long n = (long long)B_DIM * S_DIM;
  if (i >= n) return;
  int m = (int)(i / S_DIM), c = (int)(i % S_DIM);
  float core = coreF[(long long)m * NF_PAD + c];
  float d;
  if (c < A_DIM)              d = EH[(long long)m * A_DIM + c] - state[i];  // delta_h
  else if (c < A_DIM + Z_DIM) d = DL[(long long)m * Z_DIM + (c - A_DIM)];  // delta_loc
  else                        d = 0.0f;                                    // zero pad P
  out[i] = core + 0.1f * d;
}

// ---------------- host ----------------
static inline int eb(long long n) { return (int)((n + 255) / 256); }

extern "C" void kernel_launch(void* const* d_in, const int* in_sizes, int n_in,
                              void* d_out, int out_size, void* d_ws, size_t ws_size,
                              hipStream_t stream) {
  (void)in_sizes; (void)n_in; (void)out_size; (void)ws_size;
  const float* t_in      = (const float*)d_in[0];
  const float* state     = (const float*)d_in[1];
  const float* W0        = (const float*)d_in[2];
  const float* b0        = (const float*)d_in[3];
  const float* resW1     = (const float*)d_in[4];
  const float* resb1     = (const float*)d_in[5];
  const float* resW2     = (const float*)d_in[6];
  const float* resb2     = (const float*)d_in[7];
  const float* Wf        = (const float*)d_in[8];
  const float* bf_b      = (const float*)d_in[9];
  const float* lp_in_w   = (const float*)d_in[10];
  const float* lp_in_b   = (const float*)d_in[11];
  const float* lp_out_w  = (const float*)d_in[12];
  const float* lp_out_b  = (const float*)d_in[13];
  const float* ta_in_w   = (const float*)d_in[14];
  const float* ta_in_b   = (const float*)d_in[15];
  const float* ta_out_w  = (const float*)d_in[16];
  const float* ta_out_b  = (const float*)d_in[17];
  const float* loc_proj_w= (const float*)d_in[18];
  const float* loc_proj_b= (const float*)d_in[19];
  const float* loc_back_w= (const float*)d_in[20];
  const float* loc_back_b= (const float*)d_in[21];
  float* out = (float*)d_out;

  char* base = (char*)d_ws;
  size_t off = 0;
  auto take = [&](size_t bytes) -> char* {
    char* p = base + off;
    off += (bytes + 255) & ~(size_t)255;
    return p;
  };
  unsigned short* Xbf    = (unsigned short*)take((size_t)B_DIM * K0_PAD * 2);
  unsigned short* Hbf    = (unsigned short*)take((size_t)B_DIM * H_DIM * 2);
  unsigned short* Ubf    = (unsigned short*)take((size_t)B_DIM * H_DIM * 2);
  unsigned short* Sbf    = (unsigned short*)take((size_t)B_DIM * A_DIM * 2);
  unsigned short* V2bf   = (unsigned short*)take((size_t)B_DIM * A_DIM * 2);
  unsigned short* W0bf   = (unsigned short*)take((size_t)H_DIM * K0_PAD * 2);
  unsigned short* RW1bf  = (unsigned short*)take((size_t)8 * H_DIM * H_DIM * 2);
  unsigned short* RW2bf  = (unsigned short*)take((size_t)8 * H_DIM * H_DIM * 2);
  unsigned short* Wfbf   = (unsigned short*)take((size_t)NF_PAD * H_DIM * 2);
  unsigned short* taInBf = (unsigned short*)take((size_t)A_DIM * A_DIM * 2);
  unsigned short* taOutBf= (unsigned short*)take((size_t)A_DIM * A_DIM * 2);
  float* bfPad  = (float*)take((size_t)NF_PAD * 4);
  float* coreF  = (float*)take((size_t)B_DIM * NF_PAD * 4);
  float* EH     = (float*)take((size_t)B_DIM * A_DIM * 4);
  float* DL     = (float*)take((size_t)B_DIM * Z_DIM * 4);

  // ---- prep: weight conversion (bf16) + padded buffers + activation build ----
  k_pad2d  <<<eb((long long)H_DIM * K0_PAD), 256, 0, stream>>>(W0, W0bf, H_DIM, 270, H_DIM, K0_PAD);
  k_cast   <<<eb(8LL * H_DIM * H_DIM), 256, 0, stream>>>(resW1, RW1bf, 8LL * H_DIM * H_DIM);
  k_cast   <<<eb(8LL * H_DIM * H_DIM), 256, 0, stream>>>(resW2, RW2bf, 8LL * H_DIM * H_DIM);
  k_pad2d  <<<eb((long long)NF_PAD * H_DIM), 256, 0, stream>>>(Wf, Wfbf, S_DIM, H_DIM, NF_PAD, H_DIM);
  k_cast   <<<eb((long long)A_DIM * A_DIM), 256, 0, stream>>>(ta_in_w + 2 * A_DIM * A_DIM, taInBf, (long long)A_DIM * A_DIM);
  k_cast   <<<eb((long long)A_DIM * A_DIM), 256, 0, stream>>>(ta_out_w, taOutBf, (long long)A_DIM * A_DIM);
  k_padbias<<<eb(NF_PAD), 256, 0, stream>>>(bf_b, bfPad, S_DIM, NF_PAD);
  k_buildx <<<eb((long long)B_DIM * K0_PAD), 256, 0, stream>>>(state, t_in, Xbf);
  k_slice2d<<<eb((long long)B_DIM * A_DIM), 256, 0, stream>>>(state, Sbf, B_DIM, S_DIM, 0, A_DIM);

  const dim3 blk(256);
  const dim3 gH(H_DIM / 128, B_DIM / 256);   // N=1024
  const dim3 gF(NF_PAD / 128, B_DIM / 256);  // N=384
  const dim3 gA(A_DIM / 128, B_DIM / 256);   // N=256

  // ---- h = relu(x @ W0^T + b0) ----
  gemm_wmma_bf16<0><<<gH, blk, 0, stream>>>(Xbf, K0_PAD, W0bf, K0_PAD, b0,
                                            nullptr, 0, Hbf, nullptr, H_DIM, K0_PAD);
  // ---- 8 residual blocks ----
  for (int r = 0; r < 8; ++r) {
    const unsigned short* w1 = RW1bf + (size_t)r * H_DIM * H_DIM;
    const unsigned short* w2 = RW2bf + (size_t)r * H_DIM * H_DIM;
    // u = tanh(h @ W1^T + b1)
    gemm_wmma_bf16<1><<<gH, blk, 0, stream>>>(Hbf, H_DIM, w1, H_DIM, resb1 + r * H_DIM,
                                              nullptr, 0, Ubf, nullptr, H_DIM, H_DIM);
    // h = tanh(h + u @ W2^T + b2)   (in-place: each element read+written by its own thread)
    gemm_wmma_bf16<2><<<gH, blk, 0, stream>>>(Ubf, H_DIM, w2, H_DIM, resb2 + r * H_DIM,
                                              Hbf, H_DIM, Hbf, nullptr, H_DIM, H_DIM);
  }
  // ---- core = h @ Wf^T + bf  (f32, padded N) ----
  gemm_wmma_bf16<3><<<gF, blk, 0, stream>>>(Hbf, H_DIM, Wfbf, H_DIM, bfPad,
                                            nullptr, 0, nullptr, coreF, NF_PAD, H_DIM);
  // ---- ta path: v2 = h_part @ taIn^T ; EH = v2 @ taOut^T ----
  gemm_wmma_bf16<4><<<gA, blk, 0, stream>>>(Sbf, A_DIM, taInBf, A_DIM, ta_in_b + 2 * A_DIM,
                                            nullptr, 0, V2bf, nullptr, A_DIM, A_DIM);
  gemm_wmma_bf16<3><<<gA, blk, 0, stream>>>(V2bf, A_DIM, taOutBf, A_DIM, ta_out_b,
                                            nullptr, 0, nullptr, EH, A_DIM, A_DIM);
  // ---- tiny loc path ----
  k_loc<<<eb(B_DIM), 256, 0, stream>>>(state, lp_in_w, lp_in_b, lp_out_w, lp_out_b,
                                       loc_proj_w, loc_proj_b, loc_back_w, loc_back_b, DL);
  // ---- combine: out = core + 0.1*delta ----
  k_combine<<<eb((long long)B_DIM * S_DIM), 256, 0, stream>>>(coreF, EH, state, DL, out);
}
